// GraphAttentionLayer_64338610094256
// MI455X (gfx1250) — matrix operand order repaired and checked by reading, not compiled, not run on hardware
//
#include <hip/hip_runtime.h>
#include <hip/hip_bf16.h>
#include <math.h>

// ---------------------------------------------------------------------------
// GAT layer, N=2048, IN_F=512, F=64, fused attention pipeline on gfx1250 WMMA
// High/low branches fused (shared adj_un B-matrix), 64-wide column chunks.
// ---------------------------------------------------------------------------

#define N_NODES 2048
#define IN_F    512
#define F_OUT   64
#define ALPHA   0.2f

typedef _Float16 v16h __attribute__((ext_vector_type(16)));
typedef _Float16 h8v  __attribute__((ext_vector_type(8)));
typedef float    v8f  __attribute__((ext_vector_type(8)));

__device__ __forceinline__ v8f wmma_f16(v16h a, v16h b, v8f c) {
  // D = A(16x32 f16) * B(32x16 f16) + C(16x16 f32)
  return __builtin_amdgcn_wmma_f32_16x16x32_f16(
      /*neg_a=*/false, a, /*neg_b=*/false, b,
      /*c_mod=*/(short)0, c, /*reuse_a=*/false, /*reuse_b=*/false);
}

__device__ __forceinline__ float relu6f(float x) {
  return fminf(fmaxf(x, 0.f), 6.f);
}

// Build a WMMA A-tile of e = leakyrelu(s_i + d_k) on the fly.
// A-tile in this code's contiguous k order (B matches): lane holds row (lane&15); half h maps to
// K = off + h, off = (lane&16)?16:0 (caller bakes off into dk).
__device__ __forceinline__ v16h build_e(float s_i, const float* dk) {
  const float4 q0 = *(const float4*)(dk);
  const float4 q1 = *(const float4*)(dk + 4);
  const float4 q2 = *(const float4*)(dk + 8);
  const float4 q3 = *(const float4*)(dk + 12);
  const float dvs[16] = {q0.x, q0.y, q0.z, q0.w, q1.x, q1.y, q1.z, q1.w,
                         q2.x, q2.y, q2.z, q2.w, q3.x, q3.y, q3.z, q3.w};
  v16h a;
  #pragma unroll
  for (int t = 0; t < 16; ++t) {
    const float v = s_i + dvs[t];
    a[t] = (_Float16)(v > 0.f ? v : ALPHA * v);
  }
  return a;
}

// Gather a WMMA A-tile from a row-major f16 LDS row (one contiguous 16-half run).
__device__ __forceinline__ v16h gather_a(const _Float16* p) {
  const h8v pa = *(const h8v*)(p);
  const h8v pb = *(const h8v*)(p + 8);
  v16h a;
  #pragma unroll
  for (int t = 0; t < 8; ++t) { a[t] = pa[t]; a[8 + t] = pb[t]; }
  return a;
}

// --------------------------- small prep kernels ----------------------------

__global__ void gat_zero(float* csH, float* csL) {
  int t = threadIdx.x;
  if (t < 64) csH[t] = 0.f; else csL[t - 64] = 0.f;
}

// h = relu6(x @ W) for both weight matrices; also column sums via atomics.
__global__ __launch_bounds__(64) void gat_h(
    const float* __restrict__ x,
    const float* __restrict__ Wh, const float* __restrict__ Wl,
    float* __restrict__ hH, float* __restrict__ hL,
    float* __restrict__ csH, float* __restrict__ csL) {
  __shared__ float xr[IN_F];
  const int i = blockIdx.x, t = threadIdx.x;
  #pragma unroll
  for (int r = 0; r < IN_F / 64; ++r)
    xr[t + 64 * r] = x[(size_t)i * IN_F + t + 64 * r];
  __syncthreads();
  float aH = 0.f, aL = 0.f;
  for (int k = 0; k < IN_F; ++k) {
    const float xv = xr[k];
    aH += xv * Wh[k * F_OUT + t];
    aL += xv * Wl[k * F_OUT + t];
  }
  aH = relu6f(aH); aL = relu6f(aL);
  hH[(size_t)i * F_OUT + t] = aH;
  hL[(size_t)i * F_OUT + t] = aL;
  atomicAdd(csH + t, aH);
  atomicAdd(csL + t, aL);
}

// s/d attention vectors — NOTE: all four are built from h_high (matches ref).
__global__ void gat_sd(const float* __restrict__ hH,
                       const float* __restrict__ aHigh,
                       const float* __restrict__ aLow,
                       float* sH, float* dH, float* sL, float* dL) {
  const int i = blockIdx.x * blockDim.x + threadIdx.x;
  if (i >= N_NODES) return;
  float s1 = 0.f, d1 = 0.f, s2 = 0.f, d2 = 0.f;
  for (int f = 0; f < F_OUT; ++f) {
    const float h = hH[(size_t)i * F_OUT + f];
    s1 += h * aHigh[f];        d1 += h * aHigh[F_OUT + f];
    s2 += h * aLow[f];         d2 += h * aLow[F_OUT + f];
  }
  sH[i] = s1; dH[i] = d1; sL[i] = s2; dL[i] = d2;
}

// aggT[f][j] = f16( N * h[j][f] + sign * colsum[f] )   (B-matrix friendly)
__global__ void gat_aggT(const float* __restrict__ h,
                         const float* __restrict__ cs,
                         _Float16* __restrict__ aggT, float sign) {
  const int j = blockIdx.x * blockDim.x + threadIdx.x;
  const int f = blockIdx.y;
  const float v = (float)N_NODES * h[(size_t)j * F_OUT + f] + sign * cs[f];
  aggT[(size_t)f * N_NODES + j] = (_Float16)v;
}

// adjT[n][k] = f16( adj[k][n] > 0 ? 1 : 0 ) — tiled transpose+binarize.
__global__ __launch_bounds__(256) void gat_adjT(const float* __restrict__ adj,
                                                _Float16* __restrict__ adjT) {
  __shared__ float tile[32][33];
  const int tx = threadIdx.x, ty = threadIdx.y;   // 32 x 8
  const int n0 = blockIdx.x * 32, k0 = blockIdx.y * 32;
  #pragma unroll
  for (int r = 0; r < 32; r += 8)
    tile[ty + r][tx] = adj[(size_t)(k0 + ty + r) * N_NODES + n0 + tx];
  __syncthreads();
  #pragma unroll
  for (int r = 0; r < 32; r += 8) {
    const float v = tile[tx][ty + r];
    adjT[(size_t)(n0 + ty + r) * N_NODES + k0 + tx] =
        (_Float16)(v > 0.f ? 1.f : 0.f);
  }
}

// ----------------------- fused attention main kernel -----------------------
// Per block: 16 output rows, 4 waves, both branches in one pass (shared B).
// Per 64-wide column chunk:
//   Z{H,L}[16,64] = e{H,L}(s,d) @ adj_un   (stage-1 WMMA, A built on the fly)
//   P{H,L} = exp(relu6(Z)-6)               (relu6 bounds Z: no row max needed)
//   acc{H,L}[16,64] += P{H,L} @ aggT{H,L}  (stage-2 WMMA), rowsums += sum(P)
// Final: relu6(0.5*(accH/rsH + accL/rsL)).
__global__ __launch_bounds__(128) void gat_main(
    const float* __restrict__ sHp, const float* __restrict__ dHp,
    const float* __restrict__ sLp, const float* __restrict__ dLp,
    const _Float16* __restrict__ adjT,
    const _Float16* __restrict__ aggTH, const _Float16* __restrict__ aggTL,
    float* __restrict__ out) {
  __shared__ __align__(16) float dbufH[N_NODES];         // 8 KB
  __shared__ __align__(16) float dbufL[N_NODES];         // 8 KB
  __shared__ __align__(64) _Float16 pbufH[4][16][64];    // 8 KB
  __shared__ __align__(64) _Float16 pbufL[4][16][64];    // 8 KB
  __shared__ float red_acc[4][16][F_OUT];                // 16 KB
  __shared__ float red_rs[4][16];
  __shared__ float resH[16][F_OUT];                      // 4 KB

  const int lane   = threadIdx.x & 31;
  const int wave   = threadIdx.x >> 5;
  const int rowblk = blockIdx.x;
  const int colA   = lane & 15;
  const int hi     = (lane & 16) ? 1 : 0;

  for (int t = threadIdx.x; t < N_NODES; t += 128) {
    dbufH[t] = dHp[t];
    dbufL[t] = dLp[t];
  }
  __syncthreads();

  const float siH = sHp[rowblk * 16 + colA];
  const float siL = sLp[rowblk * 16 + colA];

  v8f accH[4] = {}, accL[4] = {};
  float rsH = 0.f, rsL = 0.f;

  for (int j0 = wave * 64; j0 < N_NODES; j0 += 256) {
    v8f zH[4] = {}, zL[4] = {};
    // B base: lane holds column n=j0+colA (+16t per tile), K run starts at hi*16
    const _Float16* bbase = adjT + (size_t)(j0 + colA) * N_NODES + hi * 16;
    const float*    dkH   = dbufH + hi * 16;
    const float*    dkL   = dbufL + hi * 16;

    // speculative prefetch of next chunk's B panel (drop-safe per ISA 10.5)
    {
      const int jn = (j0 + 256 < N_NODES) ? (j0 + 256) : j0;
      __builtin_prefetch(adjT + (size_t)(jn + colA) * N_NODES, 0, 1);
    }

    for (int k0 = 0; k0 < N_NODES; k0 += 32) {
      const v16h aH = build_e(siH, dkH + k0);
      const v16h aL = build_e(siL, dkL + k0);
      #pragma unroll
      for (int t = 0; t < 4; ++t) {
        const v16h b = *(const v16h*)(bbase + (size_t)(16 * t) * N_NODES + k0);
        zH[t] = wmma_f16(aH, b, zH[t]);
        zL[t] = wmma_f16(aL, b, zL[t]);
      }
    }

    // ---- P = exp(relu6(Z) - 6), stash to LDS for C->A layout change
    {
      const int r0 = hi * 8;
      #pragma unroll
      for (int t = 0; t < 4; ++t) {
        #pragma unroll
        for (int r = 0; r < 8; ++r) {
          pbufH[wave][r0 + r][16 * t + colA] =
              (_Float16)__expf(relu6f(zH[t][r]) - 6.f);
          pbufL[wave][r0 + r][16 * t + colA] =
              (_Float16)__expf(relu6f(zL[t][r]) - 6.f);
        }
      }
    }
    __syncthreads();   // cross-lane layout handoff (per-wave pbuf, in-order after)

    const _Float16* prH = &pbufH[wave][colA][0];
    const _Float16* prL = &pbufL[wave][colA][0];
    {
      float sh = 0.f, sl = 0.f;
      #pragma unroll
      for (int c = 0; c < 64; ++c) { sh += (float)prH[c]; sl += (float)prL[c]; }
      rsH += sh; rsL += sl;
    }

    // ---- stage-2: acc += P @ aggT, K in two 32-slices of the 64-wide chunk
    #pragma unroll
    for (int kk = 0; kk < 64; kk += 32) {
      const v16h a2H = gather_a(prH + kk + hi * 16);
      const v16h a2L = gather_a(prL + kk + hi * 16);
      const size_t goff = (size_t)colA * N_NODES + j0 + kk + hi * 16;
      const _Float16* gH = aggTH + goff;
      const _Float16* gL = aggTL + goff;
      #pragma unroll
      for (int t = 0; t < 4; ++t) {
        accH[t] = wmma_f16(a2H, *(const v16h*)(gH + (size_t)(16 * t) * N_NODES), accH[t]);
        accL[t] = wmma_f16(a2L, *(const v16h*)(gL + (size_t)(16 * t) * N_NODES), accL[t]);
      }
    }
  }

  // ---- 4-wave reduce: high branch -> resH, then low branch -> final output
  const int r0 = hi * 8;
  #pragma unroll
  for (int t = 0; t < 4; ++t)
    #pragma unroll
    for (int r = 0; r < 8; ++r)
      red_acc[wave][r0 + r][16 * t + colA] = accH[t][r];
  if (lane < 16) red_rs[wave][lane] = rsH;
  __syncthreads();

  #pragma unroll
  for (int e = 0; e < 8; ++e) {
    const int idx = threadIdx.x * 8 + e;
    const int row = idx >> 6, col = idx & 63;
    const float sum = red_acc[0][row][col] + red_acc[1][row][col] +
                      red_acc[2][row][col] + red_acc[3][row][col];
    const float rsum = red_rs[0][row] + red_rs[1][row] +
                       red_rs[2][row] + red_rs[3][row];
    resH[row][col] = sum / rsum;
  }
  __syncthreads();

  #pragma unroll
  for (int t = 0; t < 4; ++t)
    #pragma unroll
    for (int r = 0; r < 8; ++r)
      red_acc[wave][r0 + r][16 * t + colA] = accL[t][r];
  if (lane < 16) red_rs[wave][lane] = rsL;
  __syncthreads();

  #pragma unroll
  for (int e = 0; e < 8; ++e) {
    const int idx = threadIdx.x * 8 + e;
    const int row = idx >> 6, col = idx & 63;
    const float sum = red_acc[0][row][col] + red_acc[1][row][col] +
                      red_acc[2][row][col] + red_acc[3][row][col];
    const float rsum = red_rs[0][row] + red_rs[1][row] +
                       red_rs[2][row] + red_rs[3][row];
    const float f = 0.5f * (resH[row][col] + sum / rsum);
    out[(size_t)(rowblk * 16 + row) * F_OUT + col] = relu6f(f);
  }
}

// ------------------------------- launcher ----------------------------------

extern "C" void kernel_launch(void* const* d_in, const int* in_sizes, int n_in,
                              void* d_out, int out_size, void* d_ws, size_t ws_size,
                              hipStream_t stream) {
  (void)in_sizes; (void)n_in; (void)out_size; (void)ws_size;
  const float* input  = (const float*)d_in[0];  // [2048,512]
  const float* adj    = (const float*)d_in[1];  // [2048,2048]
  const float* Whigh  = (const float*)d_in[2];  // [512,64]
  const float* Wlow   = (const float*)d_in[3];  // [512,64]
  const float* a_high = (const float*)d_in[4];  // [128]
  const float* a_low  = (const float*)d_in[5];  // [128]
  float* out = (float*)d_out;

  char* ws = (char*)d_ws;
  size_t off = 0;
  auto take = [&](size_t bytes) -> char* {
    char* p = ws + off;
    off = (off + bytes + 255) & ~(size_t)255;
    return p;
  };
  float*    hH    = (float*)take((size_t)N_NODES * F_OUT * 4);
  float*    hL    = (float*)take((size_t)N_NODES * F_OUT * 4);
  float*    csH   = (float*)take(F_OUT * 4);
  float*    csL   = (float*)take(F_OUT * 4);
  float*    sH    = (float*)take(N_NODES * 4);
  float*    dH    = (float*)take(N_NODES * 4);
  float*    sL    = (float*)take(N_NODES * 4);
  float*    dL    = (float*)take(N_NODES * 4);
  _Float16* aggTH = (_Float16*)take((size_t)F_OUT * N_NODES * 2);
  _Float16* aggTL = (_Float16*)take((size_t)F_OUT * N_NODES * 2);
  _Float16* adjT  = (_Float16*)take((size_t)N_NODES * N_NODES * 2);

  gat_zero<<<1, 128, 0, stream>>>(csH, csL);
  gat_h<<<N_NODES, 64, 0, stream>>>(input, Whigh, Wlow, hH, hL, csH, csL);
  gat_sd<<<N_NODES / 256, 256, 0, stream>>>(hH, a_high, a_low, sH, dH, sL, dL);
  gat_aggT<<<dim3(N_NODES / 256, F_OUT), 256, 0, stream>>>(hH, csH, aggTH,  1.0f);
  gat_aggT<<<dim3(N_NODES / 256, F_OUT), 256, 0, stream>>>(hL, csL, aggTL, -1.0f);
  gat_adjT<<<dim3(N_NODES / 32, N_NODES / 32), dim3(32, 8), 0, stream>>>(adj, adjT);
  gat_main<<<N_NODES / 16, 128, 0, stream>>>(sH, dH, sL, dL, adjT, aggTH, aggTL, out);
}
